// ThinkerDecoderLayer_35046933135814
// MI455X (gfx1250) — matrix operand order repaired and checked by reading, not compiled, not run on hardware
//
#include <hip/hip_runtime.h>
#include <hip/hip_bf16.h>

// ---------------- problem constants ----------------
#define Bsz   2
#define Tsz   4096
#define Hsz   2048
#define NHsz  16
#define HDsz  128
#define INTERsz 8192
#define Kc    3           // conv kernel
#define MT    (Bsz*Tsz)   // 8192 rows
#define EPSf  1e-6f
#define BASEf 10000.0f
#define CT    128                 // scan chunk length
#define NC    (Tsz/CT)            // 32 chunks

typedef __attribute__((ext_vector_type(16))) _Float16 v16h;
typedef __attribute__((ext_vector_type(8)))  float    v8f;

// ---------------- CDNA5 async global->LDS copy (ASYNCcnt) ----------------
// Raw wave-relative LDS byte offset of a __shared__ pointer.
__device__ __forceinline__ unsigned lds_off(const void* p) {
    return (unsigned)(uintptr_t)(__attribute__((address_space(3))) const char*)p;
}
// GLOBAL_LOAD_ASYNC_TO_LDS_B128 (GV mode): LDS[vdst] = MEM[vaddr], 16 bytes/lane.
__device__ __forceinline__ void async_load_b128(unsigned lds, const void* gaddr) {
    asm volatile("global_load_async_to_lds_b128 %0, %1, off"
                 :: "v"(lds), "v"(gaddr) : "memory");
}
__device__ __forceinline__ void wait_asynccnt0() {
    asm volatile("s_wait_asynccnt 0" ::: "memory");
}

// ---------------- WMMA fragment loads from LDS ----------------
// 16-bit A-matrix 16x32 layout (ISA 7.12.2): lanes 0-15 hold K=0..7 (v0-3)
// and K=16..23 (v4-7); lanes 16-31 hold K=8..15 and K=24..31.
__device__ __forceinline__ v16h load_a_frag(const _Float16* row, int grp) {
    union { v16h v; uint4 q[2]; } u;
    u.q[0] = *reinterpret_cast<const uint4*>(row + grp * 8);
    u.q[1] = *reinterpret_cast<const uint4*>(row + 16 + grp * 8);
    return u.v;
}
// B-matrix 32x16: lanes 0-15 hold K=0..7 and K=16..23; lanes 16-31 K=8..15 and K=24..31.
__device__ __forceinline__ v16h load_b_frag(const _Float16* row, int grp) {
    union { v16h v; uint4 q[2]; } u;
    u.q[0] = *reinterpret_cast<const uint4*>(row + grp * 8);
    u.q[1] = *reinterpret_cast<const uint4*>(row + 16 + grp * 8);
    return u.v;
}

// ---------------- GEMM: C[M,N] = A[M,K] (f16, rm) * Bt[N,K] (f16) ----------
enum { EP_F32 = 0, EP_BIAS_F16 = 1, EP_RES_F32 = 2, EP_SILU_F16 = 3 };

#define BM 128
#define BN 128
#define BK 64
#define LDT 72   // padded LDS row stride (halves): 64 + 8

template <int EP>
__global__ __launch_bounds__(256, 1)
void gemm_wmma_f16(const _Float16* __restrict__ A, const _Float16* __restrict__ Bt,
                   int Kd, int Nd,
                   float* __restrict__ Cf, _Float16* __restrict__ Ch,
                   const float* __restrict__ bias, const float* __restrict__ res) {
    __shared__ _Float16 sA[2][BM * LDT];
    __shared__ _Float16 sB[2][BN * LDT];

    const int m0 = blockIdx.y * BM;
    const int n0 = blockIdx.x * BN;

    const int lane = threadIdx.x & 31;
    const int wave = threadIdx.x >> 5;
    const int wm   = wave & 3;    // 4 waves over M (32 rows each)
    const int wn   = wave >> 2;   // 2 waves over N (64 cols each)
    const int mrow = lane & 15;
    const int grp  = lane >> 4;

    const int ldrow = threadIdx.x >> 1;        // 0..127
    const int ldh   = (threadIdx.x & 1) * 32;  // column offset (halves)

    const _Float16* gA = A + (size_t)(m0 + ldrow) * Kd + ldh;
    const _Float16* gB = Bt + (size_t)(n0 + ldrow) * Kd + ldh;
    const unsigned dA0 = lds_off(&sA[0][ldrow * LDT + ldh]);
    const unsigned dB0 = lds_off(&sB[0][ldrow * LDT + ldh]);
    const unsigned bufStride = lds_off(&sA[1][0]) - lds_off(&sA[0][0]); // == BM*LDT*2

    // prologue: async-stage first K-tile into buffer 0
#pragma unroll
    for (int i = 0; i < 4; ++i) {
        async_load_b128(dA0 + i * 16, gA + i * 8);
        async_load_b128(dB0 + i * 16, gB + i * 8);
    }

    v8f acc[2][4];
#pragma unroll
    for (int i = 0; i < 2; ++i)
#pragma unroll
        for (int j = 0; j < 4; ++j) acc[i][j] = v8f{};

    int buf = 0;
    for (int k0 = 0; k0 < Kd; k0 += BK) {
        // make this wave's async LDS writes complete, then workgroup-visible
        wait_asynccnt0();
        __syncthreads();

        // overlap: issue async copies for next K-tile into the other buffer
        if (k0 + BK < Kd) {
            const unsigned nb = bufStride * (buf ^ 1);
#pragma unroll
            for (int i = 0; i < 4; ++i) {
                async_load_b128(dA0 + nb + i * 16, gA + (k0 + BK) + i * 8);
                async_load_b128(dB0 + nb + i * 16, gB + (k0 + BK) + i * 8);
            }
        }

        const _Float16* bA = sA[buf];
        const _Float16* bB = sB[buf];
#pragma unroll
        for (int kk = 0; kk < BK; kk += 32) {
            v16h af[2], bf[4];
#pragma unroll
            for (int fm = 0; fm < 2; ++fm)
                af[fm] = load_a_frag(&bA[(wm * 32 + fm * 16 + mrow) * LDT + kk], grp);
#pragma unroll
            for (int fn = 0; fn < 4; ++fn)
                bf[fn] = load_b_frag(&bB[(wn * 64 + fn * 16 + mrow) * LDT + kk], grp);
#pragma unroll
            for (int fm = 0; fm < 2; ++fm)
#pragma unroll
                for (int fn = 0; fn < 4; ++fn)
                    acc[fm][fn] = __builtin_amdgcn_wmma_f32_16x16x32_f16(
                        false, af[fm], false, bf[fn], (short)0, acc[fm][fn], false, false);
        }
        buf ^= 1;
    }

    // epilogue: C element (fm,fn,r): row = r + 8*grp, col = lane&15.
#pragma unroll
    for (int fm = 0; fm < 2; ++fm) {
#pragma unroll
        for (int fn = 0; fn < 4; ++fn) {
            const int col = n0 + wn * 64 + fn * 16 + (lane & 15);
            float bv = 0.0f;
            if (EP == EP_BIAS_F16) bv = bias[col];
#pragma unroll
            for (int r = 0; r < 8; ++r) {
                const int row = m0 + wm * 32 + fm * 16 + grp * 8 + r;
                const size_t ci = (size_t)row * Nd + col;
                float val = acc[fm][fn][r];
                if (EP == EP_F32) {
                    Cf[ci] = val;
                } else if (EP == EP_BIAS_F16) {
                    Ch[ci] = (_Float16)(val + bv);
                } else if (EP == EP_RES_F32) {
                    Cf[ci] = val + res[ci];
                } else { // EP_SILU_F16
                    float sg = 1.0f / (1.0f + __expf(-val));
                    Ch[ci] = (_Float16)(val * sg);
                }
            }
        }
    }
}

// ---------------- RMSNorm -> f16 ----------------
__global__ __launch_bounds__(256)
void rmsnorm_f16_kernel(const float* __restrict__ x, const float* __restrict__ w,
                        _Float16* __restrict__ y) {
    __shared__ float red[256];
    const size_t row = blockIdx.x;
    const float* xr = x + row * Hsz;
    float ss = 0.0f;
    for (int i = threadIdx.x; i < Hsz; i += 256) { float v = xr[i]; ss += v * v; }
    red[threadIdx.x] = ss;
    __syncthreads();
    for (int s = 128; s > 0; s >>= 1) {
        if (threadIdx.x < s) red[threadIdx.x] += red[threadIdx.x + s];
        __syncthreads();
    }
    const float scale = rsqrtf(red[0] / (float)Hsz + EPSf);
    for (int i = threadIdx.x; i < Hsz; i += 256)
        y[row * Hsz + i] = (_Float16)(xr[i] * scale * w[i]);
}

// --------- tiled transpose + f16 convert: Wt[n,k] = W[k,n] (coalesced) -----
__global__ __launch_bounds__(256)
void wtrans_tiled_kernel(const float* __restrict__ W, _Float16* __restrict__ Wt,
                         int Kd, int Nd) {
    __shared__ float tile[32][33];
    const int k0 = blockIdx.x * 32;
    const int n0 = blockIdx.y * 32;
    const int tx = threadIdx.x & 31;
    const int ty = threadIdx.x >> 5;   // 0..7
#pragma unroll
    for (int i = 0; i < 4; ++i)
        tile[ty + i * 8][tx] = W[(size_t)(k0 + ty + i * 8) * Nd + n0 + tx];
    __syncthreads();
#pragma unroll
    for (int i = 0; i < 4; ++i)
        Wt[(size_t)(n0 + ty + i * 8) * Kd + k0 + tx] = (_Float16)tile[tx][ty + i * 8];
}

// conv_w (O,I,K) -> Wt[o, k*H + i]  (Bt layout for the im2col GEMM)
__global__ void convw_kernel(const float* __restrict__ Wc, _Float16* __restrict__ Wt) {
    size_t idx = (size_t)blockIdx.x * blockDim.x + threadIdx.x;
    size_t total = (size_t)Hsz * Hsz * Kc;
    if (idx >= total) return;
    size_t o = idx / (Kc * Hsz);
    size_t rem = idx % (Kc * Hsz);
    size_t k = rem / Hsz, i = rem % Hsz;
    Wt[idx] = (_Float16)Wc[(o * Hsz + i) * Kc + k];
}

// im2col: Ac[b*T+t, k*H + i] = attn16[b, t+k-2, i]  (0 outside)
__global__ void im2col_kernel(const _Float16* __restrict__ x, _Float16* __restrict__ Ac) {
    const int CW = 3 * Hsz / 8;
    size_t tid = (size_t)blockIdx.x * blockDim.x + threadIdx.x;
    size_t total = (size_t)MT * CW;
    if (tid >= total) return;
    size_t r = tid / CW;
    int c = (int)(tid % CW) * 8;
    int k = c / Hsz, i = c % Hsz;
    int b = (int)(r / Tsz), t = (int)(r % Tsz);
    int ts = t + k - (Kc - 1);
    uint4 val = {0u, 0u, 0u, 0u};
    if (ts >= 0)
        val = *reinterpret_cast<const uint4*>(&x[((size_t)b * Tsz + ts) * Hsz + i]);
    *reinterpret_cast<uint4*>(&Ac[r * (3 * Hsz) + c]) = val;
}

// ---------------- dt projection + softplus + alpha ----------------
__global__ __launch_bounds__(256)
void dt_alpha_kernel(const _Float16* __restrict__ attn, const float* __restrict__ w_dt,
                     const float* __restrict__ b_dt, const float* __restrict__ A_log,
                     const float* __restrict__ dt_bias,
                     float* __restrict__ alpha, float* __restrict__ beta) {
    const int lane = threadIdx.x & 31;
    const int wid = blockIdx.x * 8 + (threadIdx.x >> 5);   // one wave per (row, head)
    const int row = wid / NHsz;
    const int head = wid % NHsz;
    float sum = 0.0f;
    const _Float16* ar = attn + (size_t)row * Hsz;
    for (int i = lane; i < Hsz; i += 32)
        sum += (float)ar[i] * w_dt[(size_t)i * NHsz + head];
    for (int off = 16; off > 0; off >>= 1) sum += __shfl_xor(sum, off, 32);
    if (lane == 0) {
        float xx = sum + b_dt[head] + dt_bias[head];
        float dt = (xx > 20.0f) ? xx : log1pf(expf(xx));
        float a = expf(-expf(A_log[head]) * dt);
        int b = row / Tsz, t = row % Tsz;
        size_t o = ((size_t)b * NHsz + head) * Tsz + t;
        alpha[o] = a;
        beta[o] = dt;
    }
}

// ---------------- chunked deltanet scan ----------------
// S1: p[bh,t] = cumprod(alpha) over t   (32 independent sequences)
__global__ void cumprod_kernel(const float* __restrict__ alpha, float* __restrict__ p) {
    const int bh = threadIdx.x;             // 0..B*NH-1
    float pr = 1.0f;
    const size_t base = (size_t)bh * Tsz;
    for (int t = 0; t < Tsz; ++t) { pr *= alpha[base + t]; p[base + t] = pr; }
}

// S2: chunk partial sums of u = beta*v/(p+eps)
__global__ __launch_bounds__(128)
void chunk_sum_kernel(const float* __restrict__ v, const float* __restrict__ p,
                      const float* __restrict__ beta, float* __restrict__ csum) {
    const int bh = blockIdx.x / NC;         // 0..31
    const int c  = blockIdx.x % NC;
    const int d  = threadIdx.x;             // 0..127
    const int b  = bh / NHsz, nh = bh % NHsz;
    const size_t bhT = (size_t)bh * Tsz;
    float s = 0.0f;
    for (int t = c * CT; t < (c + 1) * CT; ++t) {
        const size_t vi = ((size_t)(b * Tsz + t)) * (NHsz * HDsz) + (size_t)nh * HDsz + d;
        s += beta[bhT + t] * v[vi] / (p[bhT + t] + EPSf);
    }
    csum[((size_t)bh * NC + c) * HDsz + d] = s;
}

// S3: in-place exclusive scan of chunk sums along c
__global__ __launch_bounds__(128)
void chunk_scan_kernel(float* __restrict__ csum) {
    const int bh = blockIdx.x;
    const int d  = threadIdx.x;
    float run = 0.0f;
    for (int c = 0; c < NC; ++c) {
        const size_t i = ((size_t)bh * NC + c) * HDsz + d;
        float val = csum[i];
        csum[i] = run;
        run += val;
    }
}

// S4: final within-chunk scan + RoPE + output (f16)
__global__ __launch_bounds__(128)
void scan_out_kernel(const float* __restrict__ q, const float* __restrict__ v,
                     const float* __restrict__ p, const float* __restrict__ beta,
                     const float* __restrict__ csum, const int* __restrict__ pos_ids,
                     _Float16* __restrict__ out) {
    const int bh = blockIdx.x / NC;
    const int c  = blockIdx.x % NC;
    const int d  = threadIdx.x;
    const int b  = bh / NHsz, nh = bh % NHsz;
    const size_t bhT = (size_t)bh * Tsz;
    const float invf = __powf(BASEf, -(float)(d & 63) / 64.0f);
    const bool even = (d & 1) == 0;
    float s = csum[((size_t)bh * NC + c) * HDsz + d];
    for (int t = c * CT; t < (c + 1) * CT; ++t) {
        const size_t qi = ((size_t)(b * Tsz + t)) * (NHsz * HDsz) + (size_t)nh * HDsz + d;
        const float pr = p[bhT + t];
        s += beta[bhT + t] * v[qi] / (pr + EPSf);
        const float pos = (float)pos_ids[b * Tsz + t];
        const float th = pos * invf;
        const float qd = q[qi];
        const float qp = q[qi ^ 1];                    // rotation partner
        // x_rot[2i] = -q[2i+1] ; x_rot[2i+1] = q[2i]
        const float qr = even ? (qd * __cosf(th) - qp * __sinf(th))
                              : (qd * __cosf(th) + qp * __sinf(th));
        out[qi] = (_Float16)(qr * (pr * s));
    }
}

// ---------------- host-side launch ----------------
static constexpr size_t ALN(size_t x) { return (x + 255) & ~(size_t)255; }

extern "C" void kernel_launch(void* const* d_in, const int* in_sizes, int n_in,
                              void* d_out, int out_size, void* d_ws, size_t ws_size,
                              hipStream_t stream) {
    const float* hidden   = (const float*)d_in[0];
    const int*   pos_ids  = (const int*)d_in[1];
    const float* conv_w   = (const float*)d_in[2];
    const float* conv_b   = (const float*)d_in[3];
    const float* wq       = (const float*)d_in[4];
    const float* wv       = (const float*)d_in[5];
    const float* w_dt     = (const float*)d_in[6];
    const float* b_dt     = (const float*)d_in[7];
    const float* A_log    = (const float*)d_in[8];
    const float* dt_bias  = (const float*)d_in[9];
    const float* wo       = (const float*)d_in[10];
    const float* w1       = (const float*)d_in[11];
    const float* w2       = (const float*)d_in[12];
    const float* norm1_w  = (const float*)d_in[13];
    const float* norm2_w  = (const float*)d_in[14];
    float* outp = (float*)d_out;

    char* ws = (char*)d_ws;
    size_t off = 0;
    auto take = [&](size_t bytes) { void* pp = ws + off; off += ALN(bytes); return pp; };

    _Float16* attn16 = (_Float16*)take((size_t)MT * Hsz * 2);
    _Float16* Aconv  = (_Float16*)take((size_t)MT * 3 * Hsz * 2);
    _Float16* convwt = (_Float16*)take((size_t)Hsz * 3 * Hsz * 2);
    _Float16* x16    = (_Float16*)take((size_t)MT * Hsz * 2);
    _Float16* wqt    = (_Float16*)take((size_t)Hsz * Hsz * 2);
    _Float16* wvt    = (_Float16*)take((size_t)Hsz * Hsz * 2);
    float*    qbuf   = (float*)take((size_t)MT * Hsz * 4);
    float*    vbuf   = (float*)take((size_t)MT * Hsz * 4);
    float*    alpha  = (float*)take((size_t)Bsz * NHsz * Tsz * 4);
    float*    beta   = (float*)take((size_t)Bsz * NHsz * Tsz * 4);
    float*    pbuf   = (float*)take((size_t)Bsz * NHsz * Tsz * 4);
    float*    csum   = (float*)take((size_t)Bsz * NHsz * NC * HDsz * 4);
    _Float16* scan16 = (_Float16*)take((size_t)MT * Hsz * 2);
    _Float16* wot    = (_Float16*)take((size_t)Hsz * Hsz * 2);
    float*    hbuf   = (float*)take((size_t)MT * Hsz * 4);
    _Float16* mlp16  = (_Float16*)take((size_t)MT * Hsz * 2);
    _Float16* w1t    = (_Float16*)take((size_t)INTERsz * Hsz * 2);
    _Float16* inter16= (_Float16*)take((size_t)MT * INTERsz * 2);
    _Float16* w2t    = (_Float16*)take((size_t)Hsz * INTERsz * 2);
    (void)ws_size; (void)in_sizes; (void)n_in; (void)out_size;

    const dim3 blk256(256);

    // 1. attn_in = rmsnorm(hidden) -> f16
    rmsnorm_f16_kernel<<<MT, blk256, 0, stream>>>(hidden, norm1_w, attn16);

    // 2. weight conversion / transposition (coalesced tiled)
    {
        size_t n = (size_t)Hsz * Hsz * Kc;
        convw_kernel<<<(n + 255) / 256, blk256, 0, stream>>>(conv_w, convwt);
        wtrans_tiled_kernel<<<dim3(Hsz / 32, Hsz / 32), blk256, 0, stream>>>(wq, wqt, Hsz, Hsz);
        wtrans_tiled_kernel<<<dim3(Hsz / 32, Hsz / 32), blk256, 0, stream>>>(wv, wvt, Hsz, Hsz);
        wtrans_tiled_kernel<<<dim3(Hsz / 32, Hsz / 32), blk256, 0, stream>>>(wo, wot, Hsz, Hsz);
        wtrans_tiled_kernel<<<dim3(Hsz / 32, INTERsz / 32), blk256, 0, stream>>>(w1, w1t, Hsz, INTERsz);
        wtrans_tiled_kernel<<<dim3(INTERsz / 32, Hsz / 32), blk256, 0, stream>>>(w2, w2t, INTERsz, Hsz);
    }

    // 3. im2col for causal conv
    {
        size_t n = (size_t)MT * (3 * Hsz / 8);
        im2col_kernel<<<(n + 255) / 256, blk256, 0, stream>>>(attn16, Aconv);
    }

    // 4. conv GEMM: x = Aconv @ convW + b  (8192 x 2048 x 6144) -> f16
    gemm_wmma_f16<EP_BIAS_F16><<<dim3(Hsz / BN, MT / BM), blk256, 0, stream>>>(
        Aconv, convwt, 3 * Hsz, Hsz, nullptr, x16, conv_b, nullptr);

    // 5. q = x @ wq ; v = x @ wv  (f32 out)
    gemm_wmma_f16<EP_F32><<<dim3(Hsz / BN, MT / BM), blk256, 0, stream>>>(
        x16, wqt, Hsz, Hsz, qbuf, nullptr, nullptr, nullptr);
    gemm_wmma_f16<EP_F32><<<dim3(Hsz / BN, MT / BM), blk256, 0, stream>>>(
        x16, wvt, Hsz, Hsz, vbuf, nullptr, nullptr, nullptr);

    // 6. dt / alpha / beta
    dt_alpha_kernel<<<(MT * NHsz) / 8, blk256, 0, stream>>>(
        attn16, w_dt, b_dt, A_log, dt_bias, alpha, beta);

    // 7. chunked deltanet scan + RoPE -> f16
    cumprod_kernel<<<1, Bsz * NHsz, 0, stream>>>(alpha, pbuf);
    chunk_sum_kernel<<<Bsz * NHsz * NC, HDsz, 0, stream>>>(vbuf, pbuf, beta, csum);
    chunk_scan_kernel<<<Bsz * NHsz, HDsz, 0, stream>>>(csum);
    scan_out_kernel<<<Bsz * NHsz * NC, HDsz, 0, stream>>>(
        qbuf, vbuf, pbuf, beta, csum, pos_ids, scan16);

    // 8. h = hidden + scan @ wo
    gemm_wmma_f16<EP_RES_F32><<<dim3(Hsz / BN, MT / BM), blk256, 0, stream>>>(
        scan16, wot, Hsz, Hsz, hbuf, nullptr, nullptr, hidden);

    // 9. mlp_in = rmsnorm(h) -> f16
    rmsnorm_f16_kernel<<<MT, blk256, 0, stream>>>(hbuf, norm2_w, mlp16);

    // 10. inter = silu(mlp_in @ w1) -> f16
    gemm_wmma_f16<EP_SILU_F16><<<dim3(INTERsz / BN, MT / BM), blk256, 0, stream>>>(
        mlp16, w1t, Hsz, INTERsz, nullptr, inter16, nullptr, nullptr);

    // 11. out = h + inter @ w2
    gemm_wmma_f16<EP_RES_F32><<<dim3(Hsz / BN, MT / BM), blk256, 0, stream>>>(
        inter16, w2t, INTERsz, Hsz, outp, nullptr, nullptr, hbuf);
}
